// DecoderLayer_23252952940867
// MI455X (gfx1250) — matrix operand order repaired and checked by reading, not compiled, not run on hardware
//
#include <hip/hip_runtime.h>
#include <cstdint>

// ---------------------------------------------------------------------------
// CDNA5 (gfx1250) decoder layer: bf16 WMMA GEMMs with double-buffered
// Tensor-Data-Mover tile staging + wave-level flash attention with async
// LDS staging of V.
// Shapes fixed per reference: B=4, S=SE=2048, E=2048, H=16, D=128, F=8192.
// ---------------------------------------------------------------------------

typedef unsigned short u16;
typedef unsigned int   u32;
typedef __attribute__((ext_vector_type(16))) __bf16 v16bf;
typedef __attribute__((ext_vector_type(8)))  float  v8f;
typedef __attribute__((ext_vector_type(4)))  unsigned int v4u;
typedef __attribute__((ext_vector_type(8)))  int v8i;
typedef __attribute__((ext_vector_type(4)))  int v4i;

union Frag {
  v16bf v;
  uint4 u[2];
  u16   s[16];
};

__device__ __forceinline__ u16 f2bf(float f) {           // RNE f32 -> bf16
  u32 u = __float_as_uint(f);
  u32 r = u + 0x7FFFu + ((u >> 16) & 1u);
  return (u16)(r >> 16);
}

__device__ __forceinline__ float gelu_exact(float x) {
  return 0.5f * x * (1.0f + erff(x * 0.70710678118654752f));
}

// --- Tensor Data Mover: load a 128-row x 64-byte tile (row stride = K bf16)
// from global into LDS (contiguous 64B rows). Issued once per wave.
__device__ __forceinline__ void tdm_load_tile(u32 lds_off, const void* gptr,
                                              int k_dwords) {
  const unsigned long long ga = (unsigned long long)(uintptr_t)gptr;
  v4u g0;
  g0[0] = 1u;                                   // count=1, user descriptor
  g0[1] = lds_off;                              // lds_addr
  g0[2] = (u32)ga;                              // global_addr[31:0]
  g0[3] = (u32)((ga >> 32) & 0x1FFFFFFu) | (2u << 30);  // addr[56:32] | type=2
  v8i g1;
  g1[0] = 0x20000;                              // data_size=2 (4 bytes)
  g1[1] = (int)(16u << 16);                     // tensor_dim0 = 16 dwords
  g1[2] = (int)(128u << 16);                    // tensor_dim1 = 128 rows
  g1[3] = (int)(16u << 16);                     // tile_dim0  = 16 dwords
  g1[4] = 128;                                  // tile_dim1  = 128 rows
  g1[5] = k_dwords;                             // tensor_dim0_stride (dwords)
  g1[6] = 0;
  g1[7] = 0;
  v4i g2 = {0, 0, 0, 0};
  v4i g3 = {0, 0, 0, 0};
#if defined(__clang_major__) && (__clang_major__ >= 23)
  v8i g4 = {0, 0, 0, 0, 0, 0, 0, 0};
  __builtin_amdgcn_tensor_load_to_lds(g0, g1, g2, g3, g4, 0);
#else
  __builtin_amdgcn_tensor_load_to_lds(g0, g1, g2, g3, 0);
#endif
}

// ------------------------------- cast --------------------------------------
__global__ __launch_bounds__(256) void cast_f32_bf16(const float* __restrict__ x,
                                                     u16* __restrict__ y, int n) {
  for (int i = blockIdx.x * 256 + threadIdx.x; i < n; i += gridDim.x * 256)
    y[i] = f2bf(x[i]);
}

// ---------------------------- layernorm ------------------------------------
__global__ __launch_bounds__(256) void layernorm_to_bf16(const float* __restrict__ x,
                                                         const float* __restrict__ g,
                                                         const float* __restrict__ bb,
                                                         u16* __restrict__ y) {
  const int E = 2048;
  const int row = blockIdx.x;
  const float* xr = x + (size_t)row * E;
  __shared__ float red[256];
  const int tid = threadIdx.x;

  float s = 0.f;
  for (int i = tid; i < E; i += 256) s += xr[i];
  red[tid] = s; __syncthreads();
  for (int st = 128; st > 0; st >>= 1) { if (tid < st) red[tid] += red[tid + st]; __syncthreads(); }
  const float mean = red[0] * (1.0f / E);
  __syncthreads();

  float v = 0.f;
  for (int i = tid; i < E; i += 256) { float d = xr[i] - mean; v += d * d; }
  red[tid] = v; __syncthreads();
  for (int st = 128; st > 0; st >>= 1) { if (tid < st) red[tid] += red[tid + st]; __syncthreads(); }
  const float inv = rsqrtf(red[0] * (1.0f / E) + 1e-5f);

  u16* yr = y + (size_t)row * E;
  for (int i = tid; i < E; i += 256)
    yr[i] = f2bf((xr[i] - mean) * inv * g[i] + bb[i]);
}

// ------------------------------- GEMM --------------------------------------
// C[M,N] = A[M,K](bf16) x W[N,K](bf16)^T, f32 accumulate. K-tiles of 32 are
// DMA'd to LDS by the Tensor Data Mover, double-buffered so the DMA of tile
// k+1 overlaps the WMMAs of tile k (TDM completes in-order per wave, so
// s_wait_tensorcnt(2) releases the older A/B pair only).
// MODE 0: store bf16.  MODE 1: store f32 = Res + acc.  MODE 2: store bf16(gelu).
template <int MODE>
__global__ __launch_bounds__(256) void gemm_bf16_wmma(const u16* __restrict__ A,
                                                      const u16* __restrict__ W,
                                                      const float* __restrict__ Res,
                                                      void* __restrict__ Out,
                                                      int M, int N, int K) {
  __shared__ __align__(16) u16 sA[2][128 * 32];
  __shared__ __align__(16) u16 sB[2][128 * 32];

  const int tid   = threadIdx.x;
  const int lane  = tid & 31;
  const int wv    = tid >> 5;
  const int wr    = wv >> 1;                 // 0..3 (M direction, 32 rows each)
  const int wc    = wv & 1;                  // 0..1 (N direction, 64 cols each)
  const int tileM = blockIdx.y * 128;
  const int tileN = blockIdx.x * 128;
  const int l15   = lane & 15;
  const int hl    = lane >> 4;

  v8f acc[2][4];
#pragma unroll
  for (int i = 0; i < 2; ++i)
#pragma unroll
    for (int j = 0; j < 4; ++j)
#pragma unroll
      for (int r = 0; r < 8; ++r) acc[i][j][r] = 0.f;

  const int niter = K >> 5;
  if (tid == 0) {                            // prologue: DMA tile 0 into buf 0
    tdm_load_tile((u32)(uintptr_t)&sA[0][0], &A[(size_t)tileM * K], K >> 1);
    tdm_load_tile((u32)(uintptr_t)&sB[0][0], &W[(size_t)tileN * K], K >> 1);
  }

  for (int i = 0; i < niter; ++i) {
    const int kc  = i << 5;
    const int cur = i & 1;

    if (i + 1 < niter) {                     // stream tile i+1 behind compute
      if (tid == 0) {
        tdm_load_tile((u32)(uintptr_t)&sA[cur ^ 1][0],
                      &A[(size_t)tileM * K + kc + 32], K >> 1);
        tdm_load_tile((u32)(uintptr_t)&sB[cur ^ 1][0],
                      &W[(size_t)tileN * K + kc + 32], K >> 1);
      }
      if (kc + 64 < K) {                     // L2 hint two tiles ahead
        __builtin_prefetch(&A[(size_t)(tileM + (tid >> 1)) * K + kc + 64], 0, 0);
        __builtin_prefetch(&W[(size_t)(tileN + (tid >> 1)) * K + kc + 64], 0, 0);
      }
      __builtin_amdgcn_s_wait_tensorcnt(2);  // older A/B pair landed
    } else {
      __builtin_amdgcn_s_wait_tensorcnt(0);  // final tile
    }
    __syncthreads();                         // tile `cur` visible to all waves

    Frag a[2], b[4];
#pragma unroll
    for (int ms = 0; ms < 2; ++ms) {         // A layout: lane = M row, K halves
      const int r  = wr * 32 + ms * 16 + l15;
      const int c0 = hl * 8;
      a[ms].u[0] = *(const uint4*)&sA[cur][r * 32 + c0];
      a[ms].u[1] = *(const uint4*)&sA[cur][r * 32 + c0 + 16];
    }
#pragma unroll
    for (int ns = 0; ns < 4; ++ns) {         // B layout: lane = N col, K halves
      const int r  = wc * 64 + ns * 16 + l15;
      const int c0 = hl * 8;
      b[ns].u[0] = *(const uint4*)&sB[cur][r * 32 + c0];
      b[ns].u[1] = *(const uint4*)&sB[cur][r * 32 + c0 + 16];
    }
#pragma unroll
    for (int ms = 0; ms < 2; ++ms)
#pragma unroll
      for (int ns = 0; ns < 4; ++ns)
        acc[ms][ns] = __builtin_amdgcn_wmma_f32_16x16x32_bf16(
            false, a[ms].v, false, b[ns].v, (short)0, acc[ms][ns], false, false);

    __syncthreads();                         // all reads of buf `cur` done
  }

  // epilogue: C layout -> element (m = r + 8*hl, n = l15) per subtile
#pragma unroll
  for (int ms = 0; ms < 2; ++ms) {
#pragma unroll
    for (int ns = 0; ns < 4; ++ns) {
      const int row0 = tileM + wr * 32 + ms * 16 + 8 * hl;
      const int col  = tileN + wc * 64 + ns * 16 + l15;
#pragma unroll
      for (int r = 0; r < 8; ++r) {
        const size_t off = (size_t)(row0 + r) * N + col;
        const float vv = acc[ms][ns][r];
        if (MODE == 0)      ((u16*)Out)[off]   = f2bf(vv);
        else if (MODE == 1) ((float*)Out)[off] = Res[off] + vv;
        else                ((u16*)Out)[off]   = f2bf(gelu_exact(vv));
      }
    }
  }
}

// --------------------------- flash attention --------------------------------
// One wave per 16-row Q tile of one (b,h). Key blocks of 32, online softmax.
// V block is staged per-wave into LDS with async loads issued before the
// score WMMAs (DMA overlaps score math + softmax).
__global__ __launch_bounds__(128) void flash_attn_wmma(const u16* __restrict__ Q,
                                                       const u16* __restrict__ Kb,
                                                       const u16* __restrict__ Vb,
                                                       u16* __restrict__ O,
                                                       int Skv, int causal) {
  const int Sq = 2048, E = 2048, Dh = 128;
  const int VS = 132;                        // padded V row stride (elems)
  __shared__ float lds_p[4][16][33];         // per-wave P staging (f32 16x32)
  __shared__ __align__(16) u16 sV[4][32 * VS];

  const int lane = threadIdx.x & 31;
  const int wv   = threadIdx.x >> 5;
  const int l15  = lane & 15;
  const int hl   = lane >> 4;

  const int g  = blockIdx.x * 4 + wv;        // (b,h,qtile) flattened; Sq/16 = 128
  const int qt = g & 127;
  const int bh = g >> 7;
  const int h  = bh & 15;
  const int b  = bh >> 4;

  Frag qf[4];                                // Q tile 16x128 as 4 A-fragments
  {
    const size_t qoff = ((size_t)(b * Sq + qt * 16 + l15)) * E + h * Dh;
#pragma unroll
    for (int kc = 0; kc < 4; ++kc) {
      const int c0 = kc * 32 + hl * 8;
      qf[kc].u[0] = *(const uint4*)&Q[qoff + c0];
      qf[kc].u[1] = *(const uint4*)&Q[qoff + c0 + 16];
    }
  }

  float mrow[8], lrow[8];
  v8f o[8];
#pragma unroll
  for (int r = 0; r < 8; ++r) { mrow[r] = -1e30f; lrow[r] = 0.f; }
#pragma unroll
  for (int c = 0; c < 8; ++c)
#pragma unroll
    for (int r = 0; r < 8; ++r) o[c][r] = 0.f;

  const float scale = 0.08838834764831845f;  // 1/sqrt(128)
  const int nblk = causal ? (qt / 2 + 1) : (Skv >> 5);
  const u32 vbase = (u32)(uintptr_t)&sV[wv][0];
  const u16* sVw = &sV[wv][0];

  for (int blk = 0; blk < nblk; ++blk) {
    const int kb = blk * 32;

    // ---- async-stage V block (32 keys x 128) into per-wave LDS ----
    asm volatile("s_wait_dscnt 0x0" ::: "memory");   // prior LDS reads done
#pragma unroll
    for (int t = 0; t < 16; ++t) {
      const int key = t * 2 + hl;                    // 2 rows per instruction
      const u32 laddr = vbase + (u32)(key * VS + l15 * 8) * 2;
      const unsigned long long gaddr = (unsigned long long)(uintptr_t)
          &Vb[((size_t)(b * Skv + kb + key)) * E + h * Dh + l15 * 8];
      asm volatile("global_load_async_to_lds_b128 %0, %1, off"
                   :: "v"(laddr), "v"(gaddr) : "memory");
    }

    // ---- scores S = Q * K^T (2x 16x16 tiles), batched K-frag loads ----
    v8f sc[2];
#pragma unroll
    for (int j = 0; j < 2; ++j)
#pragma unroll
      for (int r = 0; r < 8; ++r) sc[j][r] = 0.f;

#pragma unroll
    for (int j = 0; j < 2; ++j) {
      const size_t koff = ((size_t)(b * Skv + kb + j * 16 + l15)) * E + h * Dh;
      Frag kf[4];
#pragma unroll
      for (int kc = 0; kc < 4; ++kc) {
        const int c0 = kc * 32 + hl * 8;
        kf[kc].u[0] = *(const uint4*)&Kb[koff + c0];
        kf[kc].u[1] = *(const uint4*)&Kb[koff + c0 + 16];
      }
#pragma unroll
      for (int kc = 0; kc < 4; ++kc)
        sc[j] = __builtin_amdgcn_wmma_f32_16x16x32_bf16(
            false, qf[kc].v, false, kf[kc].v, (short)0, sc[j], false, false);
    }

    // ---- scale + causal mask + online softmax; stage P (f32) in LDS ----
#pragma unroll
    for (int r = 0; r < 8; ++r) {
      float s0 = sc[0][r] * scale;
      float s1 = sc[1][r] * scale;
      if (causal) {
        const int qi = qt * 16 + r + 8 * hl;
        if (kb + l15 > qi)      s0 = -1e30f;
        if (kb + 16 + l15 > qi) s1 = -1e30f;
      }
      float t = fmaxf(s0, s1);
#pragma unroll
      for (int m = 1; m < 16; m <<= 1) t = fmaxf(t, __shfl_xor(t, m, 32));
      const float mn    = fmaxf(mrow[r], t);
      const float alpha = __expf(mrow[r] - mn);
      mrow[r] = mn;
      const float p0 = __expf(s0 - mn);
      const float p1 = __expf(s1 - mn);
      float ls = p0 + p1;
#pragma unroll
      for (int m = 1; m < 16; m <<= 1) ls += __shfl_xor(ls, m, 32);
      lrow[r] = lrow[r] * alpha + ls;
#pragma unroll
      for (int c = 0; c < 8; ++c) o[c][r] *= alpha;
      const int prow = r + 8 * hl;           // C-layout row of this vgpr/half
      lds_p[wv][prow][l15]      = p0;
      lds_p[wv][prow][16 + l15] = p1;
    }
    asm volatile("s_wait_dscnt 0x0" ::: "memory");   // intra-wave LDS ordering
    __builtin_amdgcn_wave_barrier();

    Frag pf;                                  // P as bf16 A-fragment (16x32)
    {
      const int c0 = hl * 16;
#pragma unroll
      for (int e = 0; e < 8; ++e) {
        pf.s[e]     = f2bf(lds_p[wv][l15][c0 + e]);
        pf.s[8 + e] = f2bf(lds_p[wv][l15][c0 + 8 + e]);
      }
    }

    // ---- V ready? then O += P * V (8 n-chunks of 16) ----
    asm volatile("s_wait_asynccnt 0x0" ::: "memory");
    __builtin_amdgcn_wave_barrier();
#pragma unroll
    for (int c = 0; c < 8; ++c) {
      Frag vf;                                // B-frag: lane = out col, K = keys
#pragma unroll
      for (int e = 0; e < 16; ++e)
        vf.s[e] = sVw[(hl * 16 + e) * VS + c * 16 + l15];
      o[c] = __builtin_amdgcn_wmma_f32_16x16x32_bf16(
          false, pf.v, false, vf.v, (short)0, o[c], false, false);
    }
  }

  // normalize and store merged-head output (bf16)
#pragma unroll
  for (int r = 0; r < 8; ++r) {
    const float inv = 1.0f / lrow[r];
    const int srow  = qt * 16 + r + 8 * hl;
    const size_t oo = ((size_t)(b * Sq + srow)) * E + h * Dh + l15;
#pragma unroll
    for (int c = 0; c < 8; ++c)
      O[oo + c * 16] = f2bf(o[c][r] * inv);
  }
}

// ------------------------------ launcher -----------------------------------
extern "C" void kernel_launch(void* const* d_in, const int* in_sizes, int n_in,
                              void* d_out, int out_size, void* d_ws, size_t ws_size,
                              hipStream_t stream) {
  (void)in_sizes; (void)n_in; (void)out_size; (void)ws_size;

  const float* hs    = (const float*)d_in[0];
  const float* enc   = (const float*)d_in[1];
  // d_in[2] = causal mask; applied analytically in flash_attn_wmma
  const float* wq_s  = (const float*)d_in[3];
  const float* wk_s  = (const float*)d_in[4];
  const float* wv_s  = (const float*)d_in[5];
  const float* wo_s  = (const float*)d_in[6];
  const float* wq_e  = (const float*)d_in[7];
  const float* wk_e  = (const float*)d_in[8];
  const float* wv_e  = (const float*)d_in[9];
  const float* wo_e  = (const float*)d_in[10];
  const float* w_fc1 = (const float*)d_in[11];
  const float* w_fc2 = (const float*)d_in[12];
  const float* ln1_g = (const float*)d_in[13];
  const float* ln1_b = (const float*)d_in[14];
  const float* ln2_g = (const float*)d_in[15];
  const float* ln2_b = (const float*)d_in[16];
  const float* ln3_g = (const float*)d_in[17];
  const float* ln3_b = (const float*)d_in[18];
  float* out = (float*)d_out;

  const int E = 2048, F = 8192, Btok = 4 * 2048;       // 8192 token rows
  const size_t nTok = (size_t)Btok * E;                // 16,777,216
  const size_t nFFN = (size_t)Btok * F;                // 67,108,864

  char* ws = (char*)d_ws;
  size_t off = 0;
  auto alloc = [&](size_t bytes) -> void* {
    void* p = ws + off;
    off = (off + bytes + 255) & ~(size_t)255;
    return p;
  };
  u16* bq_s  = (u16*)alloc((size_t)E * E * 2);
  u16* bk_s  = (u16*)alloc((size_t)E * E * 2);
  u16* bv_s  = (u16*)alloc((size_t)E * E * 2);
  u16* bo_s  = (u16*)alloc((size_t)E * E * 2);
  u16* bq_e  = (u16*)alloc((size_t)E * E * 2);
  u16* bk_e  = (u16*)alloc((size_t)E * E * 2);
  u16* bv_e  = (u16*)alloc((size_t)E * E * 2);
  u16* bo_e  = (u16*)alloc((size_t)E * E * 2);
  u16* bfc1  = (u16*)alloc((size_t)F * E * 2);
  u16* bfc2  = (u16*)alloc((size_t)E * F * 2);
  u16* encb  = (u16*)alloc(nTok * 2);
  u16* lnb   = (u16*)alloc(nTok * 2);
  u16* qb    = (u16*)alloc(nTok * 2);
  u16* kb    = (u16*)alloc(nTok * 2);
  u16* vb    = (u16*)alloc(nTok * 2);
  u16* ab    = (u16*)alloc(nTok * 2);
  float* h1  = (float*)alloc(nTok * 4);
  u16* fmid  = (u16*)alloc(nFFN * 2);

  auto cast = [&](const float* s, u16* d, size_t n) {
    cast_f32_bf16<<<2048, 256, 0, stream>>>(s, d, (int)n);
  };
  cast(wq_s, bq_s, (size_t)E * E);  cast(wk_s, bk_s, (size_t)E * E);
  cast(wv_s, bv_s, (size_t)E * E);  cast(wo_s, bo_s, (size_t)E * E);
  cast(wq_e, bq_e, (size_t)E * E);  cast(wk_e, bk_e, (size_t)E * E);
  cast(wv_e, bv_e, (size_t)E * E);  cast(wo_e, bo_e, (size_t)E * E);
  cast(w_fc1, bfc1, (size_t)F * E); cast(w_fc2, bfc2, (size_t)E * F);
  cast(enc, encb, nTok);

  const dim3 blk(256);
  const dim3 gridE(E / 128, Btok / 128);   // (16, 64)
  const dim3 gridF(F / 128, Btok / 128);   // (64, 64)

  // ---- self-attention ----
  layernorm_to_bf16<<<Btok, blk, 0, stream>>>(hs, ln1_g, ln1_b, lnb);
  gemm_bf16_wmma<0><<<gridE, blk, 0, stream>>>(lnb, bq_s, nullptr, qb, Btok, E, E);
  gemm_bf16_wmma<0><<<gridE, blk, 0, stream>>>(lnb, bk_s, nullptr, kb, Btok, E, E);
  gemm_bf16_wmma<0><<<gridE, blk, 0, stream>>>(lnb, bv_s, nullptr, vb, Btok, E, E);
  flash_attn_wmma<<<2048, 128, 0, stream>>>(qb, kb, vb, ab, 2048, 1);
  gemm_bf16_wmma<1><<<gridE, blk, 0, stream>>>(ab, bo_s, hs, h1, Btok, E, E);

  // ---- cross-attention ----
  layernorm_to_bf16<<<Btok, blk, 0, stream>>>(h1, ln2_g, ln2_b, lnb);
  gemm_bf16_wmma<0><<<gridE, blk, 0, stream>>>(lnb,  bq_e, nullptr, qb, Btok, E, E);
  gemm_bf16_wmma<0><<<gridE, blk, 0, stream>>>(encb, bk_e, nullptr, kb, Btok, E, E);
  gemm_bf16_wmma<0><<<gridE, blk, 0, stream>>>(encb, bv_e, nullptr, vb, Btok, E, E);
  flash_attn_wmma<<<2048, 128, 0, stream>>>(qb, kb, vb, ab, 2048, 0);
  gemm_bf16_wmma<1><<<gridE, blk, 0, stream>>>(ab, bo_e, h1, h1, Btok, E, E);

  // ---- FFN ----
  layernorm_to_bf16<<<Btok, blk, 0, stream>>>(h1, ln3_g, ln3_b, lnb);
  gemm_bf16_wmma<2><<<gridF, blk, 0, stream>>>(lnb,  bfc1, nullptr, fmid, Btok, F, E);
  gemm_bf16_wmma<1><<<gridE, blk, 0, stream>>>(fmid, bfc2, h1, out, Btok, E, F);
}
